// GraphAttentionConv_7000796693048
// MI455X (gfx1250) — matrix operand order repaired and checked by reading, not compiled, not run on hardware
//
#include <hip/hip_runtime.h>
#include <hip/hip_bf16.h>

// GAT forward for MI455X (gfx1250, wave32, WMMA).
// B=8, N=1024, C=128, H=4, F=32. Both matmuls (x@W and att@g) run on
// v_wmma_f32_16x16x32_f16; attention is fused flash-style so the (B,N,N,H)
// score tensor (134 MB) is never materialized. Adjacency + s2 tiles are
// staged into LDS with global_load_async_to_lds (ASYNCcnt path).

typedef __attribute__((ext_vector_type(16))) _Float16 v16h;
typedef __attribute__((ext_vector_type(8)))  _Float16 v8h;
typedef __attribute__((ext_vector_type(8)))  float    v8f;

#define B_  8
#define N_  1024
#define C_  128
#define H_  4
#define F_  32
#define HF_ 128

#define PW_ 136   // LDS row pitch (halfs) for 128-wide tiles: 272 B, 16 B-aligned
#define PG_ 40    // LDS row pitch (halfs) for 32-wide g tile: 80 B, 16 B-aligned

// ---- async global->LDS helpers (CDNA5; tracked with ASYNCcnt) --------------
// LDS byte offset = low 32 bits of the flat shared-aperture address (ISA 10.2).
__device__ __forceinline__ unsigned lds_off(const void* p) {
  return (unsigned)(size_t)p;
}
__device__ __forceinline__ void async_load_b128(unsigned lds, const void* g) {
  unsigned long long ga = (unsigned long long)(size_t)g;
  asm volatile("global_load_async_to_lds_b128 %0, %1, off"
               :: "v"(lds), "v"(ga) : "memory");
}
__device__ __forceinline__ void async_load_b32(unsigned lds, const void* g) {
  unsigned long long ga = (unsigned long long)(size_t)g;
  asm volatile("global_load_async_to_lds_b32 %0, %1, off"
               :: "v"(lds), "v"(ga) : "memory");
}
__device__ __forceinline__ void wait_async0() {
#if __has_builtin(__builtin_amdgcn_s_wait_asynccnt)
  __builtin_amdgcn_s_wait_asynccnt(0);
#else
  asm volatile("s_wait_asynccnt 0" ::: "memory");
#endif
}

// 16-byte aligned fragment load: 16 contiguous halfs starting at p.
__device__ __forceinline__ v16h load_frag16(const _Float16* p) {
  v8h lo = *(const v8h*)p;
  v8h hi = *(const v8h*)(p + 8);
  return __builtin_shufflevector(lo, hi, 0, 1, 2, 3, 4, 5, 6, 7,
                                 8, 9, 10, 11, 12, 13, 14, 15);
}

// ---------------------------------------------------------------------------
// Kernel 1: g = x @ W, stored f16 (read by gat_scores and as WMMA operands in gat_attn).
// Block = 128 threads (4 waves), one 16-row tile x all 128 cols. A (16x128)
// and W (128x128, transposed) are staged to LDS once; single barrier; then
// 4 K-steps of WMMA per col tile. Wave w owns col tiles 2w, 2w+1 (head w).
// ---------------------------------------------------------------------------
__global__ __launch_bounds__(128) void gat_gemm_g16(
    const float* __restrict__ x, const float* __restrict__ W,
    _Float16* __restrict__ g16)
{
  __shared__ alignas(16) _Float16 At[16 * PW_];    // [m][k]
  __shared__ alignas(16) _Float16 Wt[128 * PW_];   // [n][k] (transposed W)

  const int tid    = threadIdx.x;
  const int wave   = tid >> 5;
  const int lane   = tid & 31;
  const int laneM  = lane & 15;
  const int laneHi = lane >> 4;       // 0 or 1
  const int blk    = blockIdx.x;
  const int b      = blk / (N_ / 16);
  const int i0     = (blk % (N_ / 16)) * 16;

  // Stage A: 16x128, f32 -> f16, coalesced on k.
  for (int e = 0; e < 16; ++e) {
    int idx = e * 128 + tid;
    int m = idx >> 7, k = idx & 127;
    At[m * PW_ + k] = (_Float16)x[((size_t)b * N_ + i0 + m) * C_ + k];
  }
  // Stage W transposed: Wt[n][k] = W[k][n], coalesced on n.
  for (int e = 0; e < 128; ++e) {
    int idx = e * 128 + tid;
    int k = idx >> 7, n = idx & 127;
    Wt[n * PW_ + k] = (_Float16)W[(size_t)k * HF_ + n];
  }
  __syncthreads();

  v8f acc0 = {}, acc1 = {};
  const int n0 = (wave * 2 + 0) * 16 + laneM;
  const int n1 = (wave * 2 + 1) * 16 + laneM;

  #pragma unroll
  for (int k0 = 0; k0 < C_; k0 += 32) {
    // A fragment: lane holds row m=L%16, k = k0 + 16*laneHi + i (contig).
    const _Float16* ap = &At[laneM * PW_ + k0 + 16 * laneHi];
    v8h alo = *(const v8h*)ap;
    v8h ahi = *(const v8h*)(ap + 8);
    v16h af = __builtin_shufflevector(alo, ahi, 0, 1, 2, 3, 4, 5, 6, 7,
                                      8, 9, 10, 11, 12, 13, 14, 15);
    // B fragments: lane holds col n=L%16, k = k0 + 16*laneHi + i (contig).
    v16h bf0 = load_frag16(&Wt[n0 * PW_ + k0 + 16 * laneHi]);
    v16h bf1 = load_frag16(&Wt[n1 * PW_ + k0 + 16 * laneHi]);
    acc0 = __builtin_amdgcn_wmma_f32_16x16x32_f16(false, af, false, bf0,
                                                  (short)0, acc0, false, false);
    acc1 = __builtin_amdgcn_wmma_f32_16x16x32_f16(false, af, false, bf1,
                                                  (short)0, acc1, false, false);
  }

  // C/D layout: VGPR r, lane L -> m = r + 8*(L/16), n = L%16.
  #pragma unroll
  for (int r = 0; r < 8; ++r) {
    int m = r + 8 * laneHi;
    size_t row = ((size_t)b * N_ + i0 + m) * HF_;
    g16[row + n0] = (_Float16)acc0[r];
    g16[row + n1] = (_Float16)acc1[r];
  }
}

// ---------------------------------------------------------------------------
// Kernel 2: s1[b,n,h] = g . a[:F],  s2[b,n,h] = g . a[F:]
// (exact decomposition of attn(cat[g_i,g_j]) used by the reference)
// ---------------------------------------------------------------------------
__global__ __launch_bounds__(256) void gat_scores(
    const _Float16* __restrict__ g16, const float* __restrict__ a,
    float* __restrict__ s1, float* __restrict__ s2)
{
  int t = blockIdx.x * 256 + threadIdx.x;
  if (t >= B_ * N_ * H_) return;
  int h  = t & (H_ - 1);
  int bn = t >> 2;                      // b*N + n
  const _Float16* gp = g16 + (size_t)bn * HF_ + h * F_;
  float a1 = 0.f, a2 = 0.f;
  #pragma unroll
  for (int f = 0; f < F_; ++f) {
    float gv = (float)gp[f];
    a1 += gv * a[f];
    a2 += gv * a[F_ + f];
  }
  s1[t] = a1;
  s2[t] = a2;
}

// ---------------------------------------------------------------------------
// Kernel 3: fused masked-softmax attention aggregation (flash-style).
// Grid = B*H*(N/64) blocks of 128 threads; wave w owns i-tile of 16 rows of
// one (b,h). j consumed in blocks of 32 => K=32 => 1 WMMA per 16-col tile.
// Adjacency (64x32 ints) and s2 (32 f32) tiles are async-copied to LDS.
// ---------------------------------------------------------------------------
__global__ __launch_bounds__(128) void gat_attn(
    const _Float16* __restrict__ g16, const float* __restrict__ s1w,
    const float* __restrict__ s2w, const int* __restrict__ adj,
    float* __restrict__ out)
{
  __shared__ alignas(16) _Float16 gjT[32 * PG_];   // [f][j] == B[k][n]^T
  __shared__ alignas(16) int      adjT[64 * 32];   // [row-in-block][j-in-tile]
  __shared__ alignas(16) float    s2sh[32];

  const int tid    = threadIdx.x;
  const int wave   = tid >> 5;
  const int lane   = tid & 31;
  const int laneM  = lane & 15;
  const int laneHi = lane >> 4;
  const int blk    = blockIdx.x;
  const int n64    = blk & 15;
  const int h      = (blk >> 4) & 3;
  const int b      = blk >> 6;
  const int ibase  = n64 * 64;
  const int i0     = ibase + wave * 16;
  const int i      = i0 + laneM;      // this lane's row (duplicated L / L^16)
  const int r_lane = wave * 16 + laneM;

  const unsigned lds_adj = lds_off(&adjT[0]);
  const unsigned lds_s2  = lds_off(&s2sh[0]);

  const float s1v = s1w[((size_t)b * N_ + i) * H_ + h];
  float M = -1e30f, S = 0.f;
  v8f o0 = {}, o1 = {};

  for (int j0 = 0; j0 < N_; j0 += 32) {
    // Async-stage adjacency tile: 64 rows x 32 ints, 16 B per lane, coalesced.
    #pragma unroll
    for (int e = 0; e < 4; ++e) {
      int q  = e * 128 + tid;         // 0..511 chunks of 4 ints
      int r  = q >> 3;                // 0..63
      int c4 = (q & 7) * 4;           // 0,4,...,28
      async_load_b128(lds_adj + (unsigned)(r * 32 + c4) * 4,
                      &adj[(size_t)(ibase + r) * N_ + j0 + c4]);
    }
    if (tid < 32)
      async_load_b32(lds_s2 + tid * 4, &s2w[((size_t)b * N_ + j0 + tid) * H_ + h]);

    // Stage g_j tile transposed (f16 transpose, regular ds): coalesced on f.
    #pragma unroll
    for (int e = 0; e < 8; ++e) {
      int idx = e * 128 + tid;
      int j = idx >> 5, f = idx & 31;
      gjT[f * PG_ + j] = g16[((size_t)b * N_ + j0 + j) * HF_ + h * F_ + f];
    }
    wait_async0();
    __syncthreads();

    if (j0 + 32 < N_)
      __builtin_prefetch(&g16[((size_t)b * N_ + j0 + 32) * HF_ + h * F_], 0, 1);

    // Scores for this lane's row, iterated directly in A-fragment k order.
    float ev[16]; int msk[16];
    #pragma unroll
    for (int q = 0; q < 16; ++q) {
      int k = 16 * laneHi + q;
      int av = adjT[r_lane * 32 + k];
      float e = s1v + s2sh[k];
      e = (e > 0.f) ? e : 0.2f * e;                 // leaky_relu(0.2)
      msk[q] = av;
      ev[q]  = av ? e : -1e30f;
    }
    // Row max: local 16 + the other half-range held by lane L^16.
    float Mt = -1e30f;
    #pragma unroll
    for (int q = 0; q < 16; ++q) Mt = fmaxf(Mt, ev[q]);
    Mt = fmaxf(Mt, __shfl_xor(Mt, 16, 32));
    float newM  = fmaxf(M, Mt);
    float alpha = __expf(M - newM);

    v16h pf;
    float St = 0.f;
    #pragma unroll
    for (int q = 0; q < 16; ++q) {
      float p = msk[q] ? __expf(ev[q] - newM) : 0.f;
      St += p;
      pf[q] = (_Float16)p;
    }
    St += __shfl_xor(St, 16, 32);
    S = S * alpha + St;
    M = newM;

    // Rescale accumulators: per-row alpha broadcast (C layout m=r+8*laneHi).
    #pragma unroll
    for (int r = 0; r < 8; ++r) {
      float al = __shfl(alpha, r + 8 * laneHi, 32);
      o0[r] *= al;
      o1[r] *= al;
    }

    // B fragments: B[k][n] = g[j0+k][h*F + n] read from gjT[n][k], contig-k.
    v16h bf0 = load_frag16(&gjT[(laneM)      * PG_ + 16 * laneHi]);
    v16h bf1 = load_frag16(&gjT[(16 + laneM) * PG_ + 16 * laneHi]);
    o0 = __builtin_amdgcn_wmma_f32_16x16x32_f16(false, pf, false, bf0,
                                                (short)0, o0, false, false);
    o1 = __builtin_amdgcn_wmma_f32_16x16x32_f16(false, pf, false, bf1,
                                                (short)0, o1, false, false);
    __syncthreads();
  }

  // Normalize by row sums and store (self-loops guarantee S > 0).
  #pragma unroll
  for (int r = 0; r < 8; ++r) {
    int m = r + 8 * laneHi;
    float Sv  = __shfl(S, m, 32);
    float inv = 1.0f / Sv;
    size_t row = ((size_t)b * N_ + i0 + m) * HF_ + h * F_;
    out[row + laneM]      = o0[r] * inv;
    out[row + 16 + laneM] = o1[r] * inv;
  }
}

// ---------------------------------------------------------------------------
extern "C" void kernel_launch(void* const* d_in, const int* in_sizes, int n_in,
                              void* d_out, int out_size, void* d_ws, size_t ws_size,
                              hipStream_t stream) {
  (void)in_sizes; (void)n_in; (void)out_size; (void)ws_size;
  const float* x   = (const float*)d_in[0];   // (B,N,C)
  const float* W   = (const float*)d_in[1];   // (C,H*F)
  const float* a   = (const float*)d_in[2];   // (2F,)
  const int*   adj = (const int*)d_in[3];     // (N,N)
  float* out = (float*)d_out;                 // (B,N,H*F)

  char* ws = (char*)d_ws;
  _Float16* g16 = (_Float16*)ws;                                   // 2 MB
  float* s1 = (float*)(ws + (size_t)B_ * N_ * HF_ * sizeof(_Float16));
  float* s2 = s1 + (size_t)B_ * N_ * H_;

  gat_gemm_g16<<<B_ * N_ / 16, 128, 0, stream>>>(x, W, g16);
  gat_scores<<<(B_ * N_ * H_) / 256, 256, 0, stream>>>(g16, a, s1, s2);
  gat_attn<<<B_ * H_ * (N_ / 64), 128, 0, stream>>>(g16, s1, s2, adj, out);
}
